// MultiheadAttention_14456859918644
// MI455X (gfx1250) — matrix operand order repaired and checked by reading, not compiled, not run on hardware
//
#include <hip/hip_runtime.h>
#include <hip/hip_bf16.h>
#include <stdint.h>

#define B_  2
#define S_  2048
#define E_  1024
#define H_  16
#define DH_ 64

typedef __attribute__((ext_vector_type(16))) __bf16 v16bf;
typedef __attribute__((ext_vector_type(8)))  float  v8f;

union Frag {
  v16bf bf;
  unsigned int u[8];
  uint4 q[2];
};

// HW f32->bf16 (RNE) via __bf16 casts; lowers to v_cvt_*bf16_f32.
__device__ __forceinline__ unsigned short f2bf(float f) {
  return __builtin_bit_cast(unsigned short, (__bf16)f);
}
__device__ __forceinline__ unsigned int pack_bf16(float lo, float hi) {
  return (unsigned int)f2bf(lo) | ((unsigned int)f2bf(hi) << 16);
}

__device__ __forceinline__ v8f wmma_bf16(const Frag& a, const Frag& b, v8f c) {
  // D = A(16x32 bf16) x B(32x16 bf16) + C(16x16 f32)
  return __builtin_amdgcn_wmma_f32_16x16x32_bf16(
      /*neg_a=*/false, a.bf, /*neg_b=*/false, b.bf,
      /*c_mod=*/(short)0, c, /*reuse_a=*/false, /*reuse_b=*/false);
}

// ---------------------------------------------------------------------------
// Y[m,n] = sum_k X[m,k] * W[n,k] + bias[n]          (X:[M,K] f32, W:[N,K] f32)
// mode 0: bf16 out at [b,h,s,d]   (Q,K for attention)
// mode 1: bf16 out at [b,h,d,s]   (V transposed -> contiguous PV B-fragments)
// mode 2: f32  out row-major [m,n] (+bias) (final output projection)
// Block: 256 thr = 8 waves (2x4), tile 128(M) x 128(N), KTILE=32.
// Double-buffered LDS staging: global loads for step k+1 issued before the
// WMMAs of step k, one barrier per k-step.
// ---------------------------------------------------------------------------
__global__ __launch_bounds__(256, 2) void gemm_bias_kernel(
    const float* __restrict__ X, const float* __restrict__ W,
    const float* __restrict__ bias, void* __restrict__ out,
    int M, int N, int K, int mode)
{
  __shared__ __align__(16) unsigned short As[2 * 128 * 32];
  __shared__ __align__(16) unsigned short Bs[2 * 128 * 32];

  const int tid  = threadIdx.x;
  const int lane = tid & 31;
  const int wave = tid >> 5;
  const int lr   = lane & 15;
  const int half = lane >> 4;
  const int wm   = wave & 1;   // 2 waves along M
  const int wn   = wave >> 1;  // 4 waves along N
  const int m0   = blockIdx.y * 128;
  const int n0   = blockIdx.x * 128;

  v8f acc[4][2] = {};
  float4 ra[4], rb[4];

  // Preload k-step 0 into registers
  #pragma unroll
  for (int i = 0; i < 4; ++i) {
    int f = tid + i * 256, r = f >> 3, c4 = (f & 7) * 4;
    ra[i] = *(const float4*)(X + (size_t)(m0 + r) * K + c4);
    rb[i] = *(const float4*)(W + (size_t)(n0 + r) * K + c4);
  }

  int cur = 0;
  for (int kb = 0; kb < K; kb += 32) {
    unsigned short* A = As + cur * (128 * 32);
    unsigned short* Bsh = Bs + cur * (128 * 32);

    // Convert staged registers -> bf16 tile in LDS
    #pragma unroll
    for (int i = 0; i < 4; ++i) {
      int f = tid + i * 256, r = f >> 3, c4 = (f & 7) * 4;
      *(unsigned int*)&A[r * 32 + c4]       = pack_bf16(ra[i].x, ra[i].y);
      *(unsigned int*)&A[r * 32 + c4 + 2]   = pack_bf16(ra[i].z, ra[i].w);
      *(unsigned int*)&Bsh[r * 32 + c4]     = pack_bf16(rb[i].x, rb[i].y);
      *(unsigned int*)&Bsh[r * 32 + c4 + 2] = pack_bf16(rb[i].z, rb[i].w);
    }
    __syncthreads();

    // Issue next k-step's global loads; latency hidden behind WMMAs below
    if (kb + 32 < K) {
      #pragma unroll
      for (int i = 0; i < 4; ++i) {
        int f = tid + i * 256, r = f >> 3, c4 = (f & 7) * 4;
        ra[i] = *(const float4*)(X + (size_t)(m0 + r) * K + kb + 32 + c4);
        rb[i] = *(const float4*)(W + (size_t)(n0 + r) * K + kb + 32 + c4);
      }
    }

    // A fragments: lane holds row lane%16; word w: k=half*16+w*2
    Frag af[4];
    #pragma unroll
    for (int mi = 0; mi < 4; ++mi) {
      int row = wm * 64 + mi * 16 + lr;
      #pragma unroll
      for (int w = 0; w < 8; ++w) {
        int k = half * 16 + w * 2;
        af[mi].u[w] = *(const unsigned int*)&A[row * 32 + k];
      }
    }
    // B fragments: lane holds col lane%16; elements = 16 contiguous K at half*16
    Frag bfr[2];
    #pragma unroll
    for (int ni = 0; ni < 2; ++ni) {
      int n = wn * 32 + ni * 16 + lr;
      bfr[ni].q[0] = *(const uint4*)&Bsh[n * 32 + half * 16];
      bfr[ni].q[1] = *(const uint4*)&Bsh[n * 32 + half * 16 + 8];
    }
    #pragma unroll
    for (int mi = 0; mi < 4; ++mi)
      #pragma unroll
      for (int ni = 0; ni < 2; ++ni)
        acc[mi][ni] = wmma_bf16(af[mi], bfr[ni], acc[mi][ni]);

    cur ^= 1;
  }

  // Epilogue: C layout -> VGPR r: row = r + 8*half, col = lane%16
  #pragma unroll
  for (int ni = 0; ni < 2; ++ni) {
    int n = n0 + wn * 32 + ni * 16 + lr;
    float bv = bias[n];
    #pragma unroll
    for (int mi = 0; mi < 4; ++mi) {
      #pragma unroll
      for (int r = 0; r < 8; ++r) {
        int m = m0 + wm * 64 + mi * 16 + r + half * 8;
        float v = acc[mi][ni][r] + bv;
        if (mode == 2) {
          ((float*)out)[(size_t)m * N + n] = v;
        } else {
          int bb = m >> 11;          // m = b*S + s, S=2048
          int s  = m & (S_ - 1);
          int hh = n >> 6;           // n = h*64 + d
          int d  = n & 63;
          size_t idx = (mode == 0)
              ? ((size_t)(bb * H_ + hh) * S_ + s) * DH_ + d
              : ((size_t)(bb * H_ + hh) * DH_ + d) * S_ + s;
          ((unsigned short*)out)[idx] = f2bf(v);
        }
      }
    }
  }
}

// ---------------------------------------------------------------------------
// Fused attention: block = 4 waves cooperating on ONE 16-row query strip,
// splitting the key dimension 4 ways.
//   LDS: strip 16x2048 f32 (128KB) + per-wave O partials (16KB) + stats (1KB)
//        => ~145KB  => 2 blocks (8 waves) resident per WGP, all 4 SIMDs busy.
//   Phase 1: wave w computes scores for keys [w*512,(w+1)*512) via WMMA
//            (K B-frags loaded straight from global bf16 as 2x b128).
//   Phase 2: 8 lanes per row scan the strip; LDS-reduced row max & sum(exp).
//   Phase 3: wave w normalizes its key range, writes attn weights (single HBM
//            pass over the 537MB matrix), WMMA-accumulates partial P*V.
//   Final:   reduce 4 per-wave O partials through LDS, store attn_out.
// ---------------------------------------------------------------------------
__global__ __launch_bounds__(128) void attn_fused_kernel(
    const unsigned short* __restrict__ Qbf,   // [b,h,s,d] bf16
    const unsigned short* __restrict__ Kbf,   // [b,h,s,d] bf16
    const unsigned short* __restrict__ Vt,    // [b,h,d,s] bf16
    float* __restrict__ attnW,                // [b,h,s,s] f32
    float* __restrict__ attnOut)              // [b,s,h*64+d] f32
{
  extern __shared__ float smem[];
  float* strip = smem;                         // [16][S_]          131072 B
  float* obuf  = strip + 16 * S_;              // [4][16][DH_]       16384 B
  float* mbuf  = obuf + 4 * 16 * DH_;          // [8][16]              512 B
  float* sbuf  = mbuf + 128;                   // [8][16]              512 B

  const int tid  = threadIdx.x;
  const int lane = tid & 31;
  const int wave = tid >> 5;                   // 0..3
  const int lr   = lane & 15;
  const int half = lane >> 4;
  const int b    = blockIdx.z;
  const int h    = blockIdx.y;
  const int bh   = b * H_ + h;
  const int qbase = blockIdx.x * 16;

  const unsigned short* Qb = Qbf + (size_t)bh * S_ * DH_;
  const unsigned short* Kb = Kbf + (size_t)bh * S_ * DH_;
  const unsigned short* Vb = Vt  + (size_t)bh * DH_ * S_;

  // Q fragments (16 rows x Dh=64 -> two 16x32 A-fragments), per-wave copy
  Frag qf[2];
  #pragma unroll
  for (int c = 0; c < 2; ++c)
    #pragma unroll
    for (int w = 0; w < 8; ++w) {
      int d = c * 32 + half * 16 + w * 2;
      qf[c].u[w] = *(const unsigned int*)(Qb + (size_t)(qbase + lr) * DH_ + d);
    }

  // ---- Phase 1: wave w fills strip columns [w*512, (w+1)*512) ----
  const int j0 = wave * 32;
  for (int jj = 0; jj < 32; ++jj) {
    int j = j0 + jj;
    if (jj + 2 < 32)  // prefetch K rows two tiles ahead
      __builtin_prefetch(Kb + (size_t)((j + 2) * 16 + lr) * DH_ + half * 32, 0, 3);
    v8f sc = {};
    #pragma unroll
    for (int c = 0; c < 2; ++c) {
      Frag kf;  // B-frag: lane col = key j*16+lr; 16 contiguous d at c*32+half*16
      const unsigned short* p = Kb + (size_t)(j * 16 + lr) * DH_ + c * 32 + half * 16;
      kf.q[0] = *(const uint4*)p;
      kf.q[1] = *(const uint4*)(p + 8);
      sc = wmma_bf16(qf[c], kf, sc);
    }
    #pragma unroll
    for (int r = 0; r < 8; ++r)
      strip[(r + half * 8) * S_ + j * 16 + lr] = sc[r] * 0.125f;  // 1/sqrt(64)
  }
  __syncthreads();

  // ---- Phase 2: softmax stats; 8 segments x 16 rows over 128 lanes ----
  const int row = tid & 15;
  const int seg = tid >> 4;                    // 0..7, 256 cols each
  const float* rp = strip + row * S_ + seg * 256;
  float mx = -3.0e38f;
  for (int t = 0; t < 64; ++t) {
    float4 v = *(const float4*)(rp + t * 4);
    mx = fmaxf(mx, fmaxf(fmaxf(v.x, v.y), fmaxf(v.z, v.w)));
  }
  mbuf[seg * 16 + row] = mx;
  __syncthreads();
  float rmx = mbuf[row];
  #pragma unroll
  for (int s8 = 1; s8 < 8; ++s8) rmx = fmaxf(rmx, mbuf[s8 * 16 + row]);
  float sum = 0.f;
  for (int t = 0; t < 64; ++t) {
    float4 v = *(const float4*)(rp + t * 4);
    sum += __expf(v.x - rmx) + __expf(v.y - rmx) + __expf(v.z - rmx) + __expf(v.w - rmx);
  }
  sbuf[seg * 16 + row] = sum;
  __syncthreads();
  float rsum = 0.f;
  #pragma unroll
  for (int s8 = 0; s8 < 8; ++s8) rsum += sbuf[s8 * 16 + row];
  const float inv = 1.0f / rsum;
  // lane's row == tid&15 == lane&15 == lr, so rmx/inv line up with fragments

  // ---- Phase 3: wave w normalizes + P*V over key tiles [w*16, w*16+16) ----
  v8f oacc[4] = {};
  float* AW = attnW + ((size_t)bh * S_ + qbase) * S_;
  for (int jj = 0; jj < 16; ++jj) {
    int j2 = wave * 16 + jj;
    if (jj + 1 < 16)  // prefetch next V tile
      __builtin_prefetch(Vb + (size_t)lr * S_ + (j2 + 1) * 32 + half * 16, 0, 3);
    Frag pf;  // A-frag of probabilities, built in A layout (covers 16x32 once)
    #pragma unroll
    for (int w = 0; w < 8; ++w) {
      int col = j2 * 32 + half * 16 + w * 2;
      float2 sv = *(const float2*)&strip[lr * S_ + col];
      float p0 = __expf(sv.x - rmx) * inv;
      float p1 = __expf(sv.y - rmx) * inv;
      pf.u[w] = pack_bf16(p0, p1);
      *(float2*)&AW[(size_t)lr * S_ + col] = make_float2(p0, p1);
    }
    #pragma unroll
    for (int nt = 0; nt < 4; ++nt) {
      Frag vf;  // B-frag from transposed V: lane col = d, 16 contiguous keys
      const unsigned short* p = Vb + (size_t)(nt * 16 + lr) * S_ + j2 * 32 + half * 16;
      vf.q[0] = *(const uint4*)p;
      vf.q[1] = *(const uint4*)(p + 8);
      oacc[nt] = wmma_bf16(pf, vf, oacc[nt]);
    }
  }

  // ---- Reduce per-wave O partials via LDS and store ----
  float* ob = obuf + wave * 16 * DH_;
  #pragma unroll
  for (int nt = 0; nt < 4; ++nt)
    #pragma unroll
    for (int r = 0; r < 8; ++r)
      ob[(r + half * 8) * DH_ + nt * 16 + lr] = oacc[nt][r];
  __syncthreads();

  {
    int orow = tid >> 3;            // 0..15
    int d0   = (tid & 7) * 8;       // 0..56
    float4 lo = {0,0,0,0}, hi = {0,0,0,0};
    #pragma unroll
    for (int w = 0; w < 4; ++w) {
      const float* p = obuf + w * 16 * DH_ + orow * DH_ + d0;
      float4 a = *(const float4*)p;
      float4 c = *(const float4*)(p + 4);
      lo.x += a.x; lo.y += a.y; lo.z += a.z; lo.w += a.w;
      hi.x += c.x; hi.y += c.y; hi.z += c.z; hi.w += c.w;
    }
    float* dst = attnOut + ((size_t)b * S_ + qbase + orow) * E_ + h * DH_ + d0;
    *(float4*)dst       = lo;
    *(float4*)(dst + 4) = hi;
  }
}

// ---------------------------------------------------------------------------
extern "C" void kernel_launch(void* const* d_in, const int* in_sizes, int n_in,
                              void* d_out, int out_size, void* d_ws, size_t ws_size,
                              hipStream_t stream) {
  (void)in_sizes; (void)n_in; (void)out_size; (void)ws_size;
  const float* query = (const float*)d_in[0];
  const float* key_  = (const float*)d_in[1];
  const float* value = (const float*)d_in[2];
  const float* q_w = (const float*)d_in[3];
  const float* q_b = (const float*)d_in[4];
  const float* k_w = (const float*)d_in[5];
  const float* k_b = (const float*)d_in[6];
  const float* v_w = (const float*)d_in[7];
  const float* v_b = (const float*)d_in[8];
  const float* o_w = (const float*)d_in[9];
  const float* o_b = (const float*)d_in[10];

  float* out   = (float*)d_out;
  float* attnW = out + (size_t)B_ * S_ * E_;     // outputs concatenated flat

  // Workspace: Qbf | Kbf | Vt (bf16, 8MB each) | attnOut (f32, 16MB) = 40MB
  unsigned short* Qbf = (unsigned short*)d_ws;
  unsigned short* Kbf = Qbf + (size_t)B_ * H_ * S_ * DH_;
  unsigned short* Vt  = Kbf + (size_t)B_ * H_ * S_ * DH_;
  float* attnOut = (float*)(Vt + (size_t)B_ * H_ * S_ * DH_);

  const int M = B_ * S_, N = E_, K = E_;
  dim3 gblk(256), ggrd(N / 128, M / 128);
  hipLaunchKernelGGL(gemm_bias_kernel, ggrd, gblk, 0, stream, query, q_w, q_b, (void*)Qbf, M, N, K, 0);
  hipLaunchKernelGGL(gemm_bias_kernel, ggrd, gblk, 0, stream, key_,  k_w, k_b, (void*)Kbf, M, N, K, 0);
  hipLaunchKernelGGL(gemm_bias_kernel, ggrd, gblk, 0, stream, value, v_w, v_b, (void*)Vt,  M, N, K, 1);

  dim3 ablk(128), agrd(S_ / 16, H_, B_);
  size_t lds = (size_t)(16 * S_ + 4 * 16 * DH_ + 256) * sizeof(float);  // ~145 KB
  hipLaunchKernelGGL(attn_fused_kernel, agrd, ablk, lds, stream, Qbf, Kbf, Vt, attnW, attnOut);

  hipLaunchKernelGGL(gemm_bias_kernel, ggrd, gblk, 0, stream, attnOut, o_w, o_b, d_out, M, N, K, 2);
}
